// GGNN_26757646254514
// MI455X (gfx1250) — matrix operand order repaired and checked by reading, not compiled, not run on hardware
//
#include <hip/hip_runtime.h>
#include <hip/hip_bf16.h>

// ---------------------------------------------------------------------------
// GGNN on MI455X (gfx1250): bf16 activations + v_wmma_f32_16x16x32_bf16 GEMMs,
// f32 accumulation/state, bf16 edge-gather to halve random traffic.
// Round 2: weights repacked fragment-major so every WMMA B operand is two
// coalesced global_load_b128 instead of eight strided global_load_b32.
// ---------------------------------------------------------------------------

typedef __attribute__((ext_vector_type(16))) __bf16        v16bf;
typedef __attribute__((ext_vector_type(8)))  float         v8f;
typedef __attribute__((ext_vector_type(4)))  unsigned int  u32x4;

union AFrag { v16bf bf; u32x4 q[2]; };
union BFrag { v16bf bf; u32x4 q[2]; };

static __device__ __forceinline__ unsigned short f2bf(float f) {
  unsigned int u = __float_as_uint(f);
  u += 0x7FFFu + ((u >> 16) & 1u);           // round-to-nearest-even
  return (unsigned short)(u >> 16);
}
static __device__ __forceinline__ float bf2f(unsigned int lo16) {
  return __uint_as_float(lo16 << 16);
}
static __device__ __forceinline__ float sigm(float x) {
  return 1.0f / (1.0f + __expf(-x));
}

// B fragment, fragment-major packed: Wp[tile*256 + lane*8 + v] is exactly the
// dword that lane `lane` needs in B-operand VGPR v (K = ks*32 + (v>>2)*16 + half*8 + 2(v&3),
// pair packed little-endian; N = lane&15).  Two b128 loads per fragment.
static __device__ __forceinline__ v16bf loadBfrag(const unsigned int* __restrict__ Wp,
                                                  int tile, int lane) {
  BFrag b;
  const u32x4* p = (const u32x4*)(Wp + tile * 256 + lane * 8);
  b.q[0] = p[0];
  b.q[1] = p[1];
  return b.bf;
}

// A fragment: 16x32 bf16 A layout. Lane l (row = l&15) holds K in
// [half*8, half*8+8) and [16+half*8, 16+half*8+8): two 16B contiguous loads.
static __device__ __forceinline__ void loadA(AFrag* a, const unsigned short* row,
                                             int half) {
#pragma unroll
  for (int ks = 0; ks < 4; ++ks) {
    const unsigned short* p = row + ks * 32 + half * 8;
    a[ks].q[0] = *(const u32x4*)(p);
    a[ks].q[1] = *(const u32x4*)(p + 16);
  }
}

// ------------------------- weight packing (fragment-major) -----------------
// W [6][128][128] -> Wp [si][nt(8)][ks(4)][lane(32)][v(8)]
__global__ void k_packW(const float* __restrict__ W, unsigned int* __restrict__ Wp) {
  int idx = blockIdx.x * 256 + threadIdx.x;            // [0, 6*8*4*256)
  if (idx >= 6 * 8 * 4 * 256) return;
  int v    = idx & 7;
  int lane = (idx >> 3) & 31;
  int ks   = (idx >> 8) & 3;
  int nt   = (idx >> 10) & 7;
  int si   = idx >> 13;
  int k = ks * 32 + (v >> 2) * 16 + (lane >> 4) * 8 + 2 * (v & 3);
  int n = nt * 16 + (lane & 15);
  int in0 = si * 16384 + k * 128 + n;                  // W[si][k][n]
  unsigned int lo = f2bf(W[in0]);
  unsigned int hi = f2bf(W[in0 + 128]);                // W[si][k+1][n]
  Wp[idx] = lo | (hi << 16);
}

// wih/whh [3][384][128] (B[k][j] = Wg[j][k]) -> Gp [s][jt(24)][ks(4)][lane][v]
__global__ void k_packG(const float* __restrict__ Wg, unsigned int* __restrict__ Gp) {
  int idx = blockIdx.x * 256 + threadIdx.x;            // [0, 3*24*4*256)
  if (idx >= 3 * 24 * 4 * 256) return;
  int v    = idx & 7;
  int lane = (idx >> 3) & 31;
  int ks   = (idx >> 8) & 3;
  int rem  = idx >> 10;                                // s*24 + jt
  int jt   = rem % 24;
  int s    = rem / 24;
  int k = ks * 32 + (v >> 2) * 16 + (lane >> 4) * 8 + 2 * (v & 3);
  int j = jt * 16 + (lane & 15);
  int in0 = (s * 384 + j) * 128 + k;                   // Wg[j][k]
  unsigned int lo = f2bf(Wg[in0]);
  unsigned int hi = f2bf(Wg[in0 + 1]);                 // Wg[j][k+1]
  Gp[idx] = lo | (hi << 16);
}

// ------------------------- elementwise helpers -----------------------------
__global__ void k_init(const float* __restrict__ x, float* __restrict__ xf,
                       unsigned short* __restrict__ xb, int NT) {
  int i = blockIdx.x * 256 + threadIdx.x;
  if (i >= NT) return;
  float v = x[i];
  xf[i] = v;
  xb[i] = f2bf(v);
}

__global__ void k_count(const int* __restrict__ dst, const int* __restrict__ attr,
                        float* __restrict__ cnt, int N, int E) {
  int e = blockIdx.x * 256 + threadIdx.x;
  if (e >= E) return;
  atomicAdd(&cnt[attr[e] * N + dst[e]], 1.0f);
}

__global__ void k_invcnt(float* __restrict__ cnt, int n) {
  int i = blockIdx.x * 256 + threadIdx.x;
  if (i >= n) return;
  cnt[i] = 1.0f / fmaxf(cnt[i], 1.0f);
}

__global__ void k_divconv(const float* __restrict__ agg, const float* __restrict__ inv,
                          unsigned short* __restrict__ aggb, int NT) {
  int i = blockIdx.x * 256 + threadIdx.x;
  if (i >= NT) return;
  aggb[i] = f2bf(agg[i] * inv[i >> 7]);
}

__global__ void k_accadd(float* __restrict__ acc, const float* __restrict__ h, int NT) {
  int i = blockIdx.x * 256 + threadIdx.x;
  if (i >= NT) return;
  acc[i] += h[i];
}

__global__ void k_xupdate(const float* __restrict__ acc, float* __restrict__ xf,
                          unsigned short* __restrict__ xb, int NT) {
  int i = blockIdx.x * 256 + threadIdx.x;
  if (i >= NT) return;
  float v = acc[i] * (1.0f / 3.0f);
  xf[i] = v;
  xb[i] = f2bf(v);
}

// ------------------------- m = h @ W[s,i]  (WMMA) --------------------------
__global__ __launch_bounds__(256) void k_gemm_m(const unsigned short* __restrict__ hb,
                                                const unsigned int* __restrict__ Wp,
                                                unsigned short* __restrict__ mb,
                                                int N) {
  const int lane = threadIdx.x & 31;
  const int wave = threadIdx.x >> 5;
  const int col  = lane & 15;
  const int half = lane >> 4;
  const int rowbase = (blockIdx.x * 8 + wave) * 16;
  if (rowbase >= N) return;                 // wave-uniform: EXEC stays all-1s

  AFrag a[4];
  loadA(a, hb + (size_t)(rowbase + col) * 128, half);

#pragma unroll
  for (int nt = 0; nt < 8; ++nt) {
    v8f c = {0.f, 0.f, 0.f, 0.f, 0.f, 0.f, 0.f, 0.f};
#pragma unroll
    for (int ks = 0; ks < 4; ++ks) {
      v16bf b = loadBfrag(Wp, nt * 4 + ks, lane);
      c = __builtin_amdgcn_wmma_f32_16x16x32_bf16(false, a[ks].bf, false, b,
                                                  (short)0, c, false, false);
    }
#pragma unroll
    for (int v = 0; v < 8; ++v)             // C/D: M = half*8 + v, N = col
      mb[(size_t)(rowbase + half * 8 + v) * 128 + nt * 16 + col] = f2bf(c[v]);
  }
}

// ------------------------- edge scatter (mean aggr numerator) --------------
__global__ void k_scatter(const unsigned short* __restrict__ mb,
                          const int* __restrict__ src, const int* __restrict__ dst,
                          const int* __restrict__ attr, int s,
                          float* __restrict__ agg, int E) {
  int idx = blockIdx.x * 256 + threadIdx.x;   // one wave per edge: divergence-free
  int e = idx >> 5;
  if (e >= E || attr[e] != s) return;
  int c4 = (idx & 31) << 2;
  const unsigned int* ms = (const unsigned int*)(mb + (size_t)src[e] * 128 + c4);
  float* ag = agg + (size_t)dst[e] * 128 + c4;
  unsigned int u0 = ms[0], u1 = ms[1];
  atomicAdd(ag + 0, bf2f(u0 & 0xffffu));
  atomicAdd(ag + 1, bf2f(u0 >> 16));
  atomicAdd(ag + 2, bf2f(u1 & 0xffffu));
  atomicAdd(ag + 3, bf2f(u1 >> 16));
}

// ------------------------- fused GRU cell (two WMMA GEMMs + gates) ---------
__global__ __launch_bounds__(256) void k_gru(const unsigned short* __restrict__ aggb,
                                             const unsigned short* hbin,
                                             float* hf,               // in/out
                                             unsigned short* hbout,   // may alias hbin
                                             const unsigned int* __restrict__ Wihp,
                                             const unsigned int* __restrict__ Whhp,
                                             const float* __restrict__ bih,
                                             const float* __restrict__ bhh,
                                             int N) {
  const int lane = threadIdx.x & 31;
  const int wave = threadIdx.x >> 5;
  const int col  = lane & 15;
  const int half = lane >> 4;
  const int rowbase = (blockIdx.x * 8 + wave) * 16;
  if (rowbase >= N) return;

  AFrag aA[4], aH[4];
  loadA(aA, aggb + (size_t)(rowbase + col) * 128, half);
  loadA(aH, hbin + (size_t)(rowbase + col) * 128, half);

  const v8f z8 = {0.f, 0.f, 0.f, 0.f, 0.f, 0.f, 0.f, 0.f};
#pragma unroll 1
  for (int c = 0; c < 8; ++c) {             // 16-column chunk of D
    v8f ir = z8, iz = z8, inn = z8, hr = z8, hz = z8, hn = z8;
#pragma unroll
    for (int ks = 0; ks < 4; ++ks) {
      v16bf b;
      b = loadBfrag(Wihp, (0 * 8 + c) * 4 + ks, lane);
      ir = __builtin_amdgcn_wmma_f32_16x16x32_bf16(false, aA[ks].bf, false, b, (short)0, ir, false, false);
      b = loadBfrag(Wihp, (1 * 8 + c) * 4 + ks, lane);
      iz = __builtin_amdgcn_wmma_f32_16x16x32_bf16(false, aA[ks].bf, false, b, (short)0, iz, false, false);
      b = loadBfrag(Wihp, (2 * 8 + c) * 4 + ks, lane);
      inn = __builtin_amdgcn_wmma_f32_16x16x32_bf16(false, aA[ks].bf, false, b, (short)0, inn, false, false);
      b = loadBfrag(Whhp, (0 * 8 + c) * 4 + ks, lane);
      hr = __builtin_amdgcn_wmma_f32_16x16x32_bf16(false, aH[ks].bf, false, b, (short)0, hr, false, false);
      b = loadBfrag(Whhp, (1 * 8 + c) * 4 + ks, lane);
      hz = __builtin_amdgcn_wmma_f32_16x16x32_bf16(false, aH[ks].bf, false, b, (short)0, hz, false, false);
      b = loadBfrag(Whhp, (2 * 8 + c) * 4 + ks, lane);
      hn = __builtin_amdgcn_wmma_f32_16x16x32_bf16(false, aH[ks].bf, false, b, (short)0, hn, false, false);
    }
    const int colb = c * 16 + col;
    const float bri = bih[colb],       brh = bhh[colb];
    const float bzi = bih[128 + colb], bzh = bhh[128 + colb];
    const float bni = bih[256 + colb], bnh = bhh[256 + colb];
#pragma unroll
    for (int v = 0; v < 8; ++v) {
      const size_t o = (size_t)(rowbase + half * 8 + v) * 128 + colb;
      const float hv = hf[o];
      const float r  = sigm(ir[v] + bri + hr[v] + brh);
      const float z  = sigm(iz[v] + bzi + hz[v] + bzh);
      const float nn = tanhf(inn[v] + bni + r * (hn[v] + bnh));
      const float hnew = (1.0f - z) * nn + z * hv;
      hf[o] = hnew;
      hbout[o] = f2bf(hnew);
    }
  }
}

// ------------------------- pooling + MLP head ------------------------------
__global__ void k_pool(const float* __restrict__ xf, const int* __restrict__ batch,
                       float* __restrict__ pooled, float* __restrict__ gcnt, int NT) {
  int i = blockIdx.x * 256 + threadIdx.x;
  if (i >= NT) return;
  int n = i >> 7, d = i & 127;
  int g = batch[n];
  atomicAdd(&pooled[g * 128 + d], xf[i]);
  if (d == 0) atomicAdd(&gcnt[g], 1.0f);
}

__global__ void k_mlp(const float* __restrict__ pooled, const float* __restrict__ gcnt,
                      const float* __restrict__ pt,
                      const float* __restrict__ fc1w, const float* __restrict__ fc1b,
                      const float* __restrict__ fc2w, const float* __restrict__ fc2b,
                      const float* __restrict__ fclw, const float* __restrict__ fclb,
                      float* __restrict__ out, int G) {
  int g = blockIdx.x * blockDim.x + threadIdx.x;
  if (g >= G) return;
  float in[129], a1[80], a2[80];
  float inv = 1.0f / fmaxf(gcnt[g], 1.0f);
  for (int d = 0; d < 128; ++d) in[d] = pooled[g * 128 + d] * inv;
  in[128] = pt[g];
  for (int o = 0; o < 80; ++o) {
    float s = fc1b[o];
    for (int k = 0; k < 129; ++k) s += fc1w[o * 129 + k] * in[k];
    a1[o] = s > 0.f ? s : 0.01f * s;
  }
  for (int o = 0; o < 80; ++o) {
    float s = fc2b[o];
    for (int k = 0; k < 80; ++k) s += fc2w[o * 80 + k] * a1[k];
    a2[o] = s > 0.f ? s : 0.01f * s;
  }
  for (int o = 0; o < 2; ++o) {
    float s = fclb[o];
    for (int k = 0; k < 80; ++k) s += fclw[o * 80 + k] * a2[k];
    out[g * 2 + o] = s;
  }
}

// ---------------------------------------------------------------------------
extern "C" void kernel_launch(void* const* d_in, const int* in_sizes, int n_in,
                              void* d_out, int out_size, void* d_ws, size_t ws_size,
                              hipStream_t stream) {
  (void)n_in; (void)out_size; (void)ws_size;
  const float* x    = (const float*)d_in[0];
  const int*   ei   = (const int*)d_in[1];
  const int*   attr = (const int*)d_in[2];
  const int*   batc = (const int*)d_in[3];
  const float* pt   = (const float*)d_in[4];
  const float* W    = (const float*)d_in[5];
  const float* wih  = (const float*)d_in[6];
  const float* whh  = (const float*)d_in[7];
  const float* bih  = (const float*)d_in[8];
  const float* bhh  = (const float*)d_in[9];
  const float* fc1w = (const float*)d_in[10];
  const float* fc1b = (const float*)d_in[11];
  const float* fc2w = (const float*)d_in[12];
  const float* fc2b = (const float*)d_in[13];
  const float* fclw = (const float*)d_in[14];
  const float* fclb = (const float*)d_in[15];
  float* out = (float*)d_out;

  const int N = in_sizes[0] / 128;          // 100000 (divisible by 16)
  const int E = in_sizes[2];                // 600000
  const int G = in_sizes[4];                // 64
  const int* src = ei;
  const int* dst = ei + E;
  const int NT = N * 128;

  // workspace carve-out (256B aligned)
  char* ws = (char*)d_ws;
  size_t off = 0;
  auto take = [&](size_t bytes) -> char* {
    char* p = ws + off;
    off = (off + bytes + 255) & ~(size_t)255;
    return p;
  };
  float*          xf   = (float*)take((size_t)NT * 4);
  float*          hf   = (float*)take((size_t)NT * 4);
  float*          agg  = (float*)take((size_t)NT * 4);
  float*          acc  = (float*)take((size_t)NT * 4);
  unsigned short* xb   = (unsigned short*)take((size_t)NT * 2);
  unsigned short* hb   = (unsigned short*)take((size_t)NT * 2);
  unsigned short* mb   = (unsigned short*)take((size_t)NT * 2);
  unsigned short* aggb = (unsigned short*)take((size_t)NT * 2);
  float*          invc = (float*)take((size_t)3 * N * 4);
  unsigned int*   Wp   = (unsigned int*)take((size_t)6 * 8 * 4 * 256 * 4);
  unsigned int*   Wihp = (unsigned int*)take((size_t)3 * 24 * 4 * 256 * 4);
  unsigned int*   Whhp = (unsigned int*)take((size_t)3 * 24 * 4 * 256 * 4);
  float*          pooled = (float*)take((size_t)G * 128 * 4);
  float*          gcnt   = (float*)take((size_t)G * 4);

  const int eltGrid  = (NT + 255) / 256;
  const int tileGrid = (N + 127) / 128;     // 128 rows / block (8 waves x 16)

  // one-time prep (repeated each call: deterministic, cheap)
  k_packW<<<(6 * 8 * 4 * 256 + 255) / 256, 256, 0, stream>>>(W, Wp);
  k_packG<<<(3 * 24 * 4 * 256 + 255) / 256, 256, 0, stream>>>(wih, Wihp);
  k_packG<<<(3 * 24 * 4 * 256 + 255) / 256, 256, 0, stream>>>(whh, Whhp);

  hipMemsetAsync(invc, 0, (size_t)3 * N * 4, stream);
  k_count<<<(E + 255) / 256, 256, 0, stream>>>(dst, attr, invc, N, E);
  k_invcnt<<<(3 * N + 255) / 256, 256, 0, stream>>>(invc, 3 * N);

  k_init<<<eltGrid, 256, 0, stream>>>(x, xf, xb, NT);

  for (int pass = 0; pass < 2; ++pass) {
    hipMemsetAsync(acc, 0, (size_t)NT * 4, stream);
    for (int s = 0; s < 3; ++s) {
      hipMemcpyAsync(hf, xf, (size_t)NT * 4, hipMemcpyDeviceToDevice, stream);
      hipMemcpyAsync(hb, xb, (size_t)NT * 2, hipMemcpyDeviceToDevice, stream);
      for (int i = 0; i < 2; ++i) {
        k_gemm_m<<<tileGrid, 256, 0, stream>>>(hb, Wp + (s * 2 + i) * 8192, mb, N);
        hipMemsetAsync(agg, 0, (size_t)NT * 4, stream);
        k_scatter<<<(E * 32 + 255) / 256, 256, 0, stream>>>(mb, src, dst, attr, s, agg, E);
        k_divconv<<<eltGrid, 256, 0, stream>>>(agg, invc + s * N, aggb, NT);
        k_gru<<<tileGrid, 256, 0, stream>>>(aggb, hb, hf, hb,
                                            Wihp + s * 24576, Whhp + s * 24576,
                                            bih + s * 384, bhh + s * 384, N);
      }
      k_accadd<<<eltGrid, 256, 0, stream>>>(acc, hf, NT);
    }
    k_xupdate<<<eltGrid, 256, 0, stream>>>(acc, xf, xb, NT);
  }

  hipMemsetAsync(pooled, 0, (size_t)G * 128 * 4, stream);
  hipMemsetAsync(gcnt, 0, (size_t)G * 4, stream);
  k_pool<<<eltGrid, 256, 0, stream>>>(xf, batc, pooled, gcnt, NT);
  k_mlp<<<1, 64, 0, stream>>>(pooled, gcnt, pt, fc1w, fc1b, fc2w, fc2b, fclw, fclb, out, G);
}
